// SemanticOnlyModel_35588099014943
// MI455X (gfx1250) — compile-verified, matrix operand order checked by reading, not run on hardware
//
#include <hip/hip_runtime.h>
#include <hip/hip_bf16.h>

// Problem constants (match reference)
#define VV   32000
#define H    256
#define H2   512
#define HALF 128
#define BB   64
#define LL   2048
#define TM   16     // tokens per encoder workgroup

typedef __attribute__((ext_vector_type(16))) _Float16 v16h;
typedef __attribute__((ext_vector_type(8)))  float    v8f;

// ---------------------------------------------------------------------------
// Fragment loader: 16x32 f16 tile, row-major source with leading dim `ld`.
// Matches CDNA5 16-bit A-matrix layout (CDNA5 ISA 7.12.2):
//   lane 0-15: M=lane,  K base 0 ; lane 16-31: M=lane-16, K base +8
//   VGPR v in 0..3: K = 2v,2v+1 ; v in 4..7: K = 16+2(v-4), +1
// B operand uses the mirrored layout (row = N), so we feed pre-transposed
// (N-major) weights through the same loader.
// ---------------------------------------------------------------------------
__device__ __forceinline__ v16h frag_ld(const _Float16* __restrict__ base,
                                        int ld, int lane) {
  const int r  = lane & 15;
  const int hi = lane >> 4;
  const _Float16* row = base + (long)r * ld + hi * 8;
  v16h f;
#pragma unroll
  for (int v = 0; v < 8; ++v) {
    const int kb = ((v & 4) << 2) + 2 * (v & 3);   // 0,2,4,6,16,18,20,22
    f[2 * v]     = row[kb];
    f[2 * v + 1] = row[kb + 1];
  }
  return f;
}

__device__ __forceinline__ v8f wmma_f16(v16h a, v16h b, v8f c) {
  return __builtin_amdgcn_wmma_f32_16x16x32_f16(false, a, false, b,
                                                (short)0, c, false, false);
}

// ---------------------------------------------------------------------------
// Kernel 0: convert f32 [K][N] weight -> f16 [N][K] (transposed, N-major)
// ---------------------------------------------------------------------------
__global__ void cvt_t_kernel(const float* __restrict__ src,
                             _Float16* __restrict__ dst, int K, int N) {
  long idx = (long)blockIdx.x * blockDim.x + threadIdx.x;
  long total = (long)K * N;
  if (idx < total) {
    int n = (int)(idx % N);
    int k = (int)(idx / N);
    dst[(long)n * K + k] = (_Float16)src[idx];
  }
}

// ---------------------------------------------------------------------------
// Kernel 1: encoder. One 256-thread block (8 wave32) handles TM=16 tokens:
//   gather embed -> GEMM1+ReLU -> GEMM2 + residual -> LayerNorm -> GEMM3 (Wp)
// All GEMMs via v_wmma_f32_16x16x32_f16.
// ---------------------------------------------------------------------------
__global__ __launch_bounds__(256) void encoder_kernel(
    const int* __restrict__ seq, const float* __restrict__ embed,
    const _Float16* __restrict__ W1t, const float* __restrict__ b1,
    const _Float16* __restrict__ W2t, const float* __restrict__ b2,
    const float* __restrict__ gamma, const float* __restrict__ beta,
    const _Float16* __restrict__ Wpt, const float* __restrict__ bp,
    float* __restrict__ p_out) {
  __shared__ int      s_tok[TM];
  __shared__ float    s_e[TM][H];        // e (f32), becomes x = e + f
  __shared__ _Float16 s_a[TM][H];        // f16 A operand (e, later h)
  __shared__ _Float16 s_c1[TM][H2];      // relu(e@W1+b1) in f16
  __shared__ float    s_red[2][TM][16];  // LN partial sums

  const int  tid      = threadIdx.x;
  const int  lane     = tid & 31;
  const int  w        = tid >> 5;                 // wave id 0..7
  const long tok_base = (long)blockIdx.x * TM;

  if (tid < TM) s_tok[tid] = seq[tok_base + tid];
  __syncthreads();

  // gather embedding rows (L2-resident table), keep f32 for residual
  for (int i = tid; i < TM * H; i += 256) {
    int r = i >> 8;           // / H
    int c = i & (H - 1);
    float v = embed[(long)s_tok[r] * H + c];
    s_e[r][c] = v;
    s_a[r][c] = (_Float16)v;
  }
  __syncthreads();

  // ---- GEMM1: [16 x 512] = A[16x256] @ W1, +b1, ReLU -> s_c1 (f16) ----
#pragma unroll
  for (int i = 0; i < 4; ++i) {
    const int n0 = (w * 4 + i) * 16;              // 32 N-tiles / 8 waves
    v8f acc = {};
    for (int k0 = 0; k0 < H; k0 += 32) {
      v16h a = frag_ld(&s_a[0][0] + k0, H, lane);
      v16h b = frag_ld(W1t + (long)n0 * H + k0, H, lane);
      acc = wmma_f16(a, b, acc);
    }
    const int lo = lane & 15, hi = lane >> 4;
    const float bias = b1[n0 + lo];
#pragma unroll
    for (int g = 0; g < 8; ++g) {
      const int r = g + hi * 8;
      float v = acc[g] + bias;
      s_c1[r][n0 + lo] = (_Float16)(v > 0.f ? v : 0.f);
    }
  }
  __syncthreads();

  // ---- GEMM2: f = C1[16x512] @ W2, +b2 ; x = e + f (into s_e) ----
#pragma unroll
  for (int i = 0; i < 2; ++i) {
    const int n0 = (w + i * 8) * 16;              // 16 N-tiles / 8 waves
    v8f acc = {};
    for (int k0 = 0; k0 < H2; k0 += 32) {
      v16h a = frag_ld(&s_c1[0][0] + k0, H2, lane);
      v16h b = frag_ld(W2t + (long)n0 * H2 + k0, H2, lane);
      acc = wmma_f16(a, b, acc);
    }
    const int lo = lane & 15, hi = lane >> 4;
    const float bias = b2[n0 + lo];
#pragma unroll
    for (int g = 0; g < 8; ++g) {
      const int r = g + hi * 8;
      s_e[r][n0 + lo] += acc[g] + bias;
    }
  }
  __syncthreads();

  // ---- LayerNorm over H=256 per row; h -> s_a (f16) ----
  {
    const int r = tid >> 4;       // 0..15
    const int part = tid & 15;    // 16 threads per row
    float s = 0.f, ss = 0.f;
#pragma unroll
    for (int j = 0; j < 16; ++j) {
      float v = s_e[r][part * 16 + j];
      s += v; ss += v * v;
    }
    s_red[0][r][part] = s;
    s_red[1][r][part] = ss;
    __syncthreads();
    float mu = 0.f, m2 = 0.f;
#pragma unroll
    for (int j = 0; j < 16; ++j) { mu += s_red[0][r][j]; m2 += s_red[1][r][j]; }
    mu *= (1.f / H);
    float var = m2 * (1.f / H) - mu * mu;
    float rstd = rsqrtf(var + 1e-5f);
#pragma unroll
    for (int j = 0; j < 16; ++j) {
      int c = part * 16 + j;
      float hh = (s_e[r][c] - mu) * rstd * gamma[c] + beta[c];
      s_a[r][c] = (_Float16)hh;
    }
  }
  __syncthreads();

  // ---- GEMM3: p[16x128] = h @ Wp, +bp -> global f32 ----
  {
    const int n0 = w * 16;                        // 8 N-tiles / 8 waves
    v8f acc = {};
    for (int k0 = 0; k0 < H; k0 += 32) {
      v16h a = frag_ld(&s_a[0][0] + k0, H, lane);
      v16h b = frag_ld(Wpt + (long)n0 * H + k0, H, lane);
      acc = wmma_f16(a, b, acc);
    }
    const int lo = lane & 15, hi = lane >> 4;
    const float bias = bp[n0 + lo];
#pragma unroll
    for (int g = 0; g < 8; ++g) {
      const int r = g + hi * 8;
      p_out[(tok_base + r) * HALF + n0 + lo] = acc[g] + bias;
    }
  }
}

// ---------------------------------------------------------------------------
// Kernel 2: delta-rule recurrence. One 512-thread block per batch element.
// M[128][128] register-resident: thread (row = tid&127, quarter q = tid>>7)
// owns columns q*32..q*32+31 of its row (32 f32 regs).
// ---------------------------------------------------------------------------
__global__ __launch_bounds__(512) void delta_kernel(
    const float* __restrict__ p, _Float16* __restrict__ cs_h) {
  __shared__ float kLDS[HALF];
  __shared__ float parts[512];
  __shared__ float dvs[HALF];
  __shared__ float kkLDS;

  const int tid = threadIdx.x;
  const int row = tid & (HALF - 1);
  const int cb  = (tid >> 7) * 32;
  float Mreg[32];
#pragma unroll
  for (int j = 0; j < 32; ++j) Mreg[j] = 0.f;

  const float* pb = p + (long)blockIdx.x * LL * HALF;

  for (int t = 0; t < LL - 1; ++t) {
    if (tid < HALF) kLDS[tid] = pb[(long)t * HALF + tid];
    __syncthreads();
    // partial matvec: vps = M @ k
    float part = 0.f;
#pragma unroll
    for (int j = 0; j < 32; ++j) part += Mreg[j] * kLDS[cb + j];
    parts[tid] = part;
    // wave 0 reduces k.k via cross-lane shuffles
    if (tid < 32) {
      float s = 0.f;
#pragma unroll
      for (int i = 0; i < 4; ++i) { float kv = kLDS[tid + 32 * i]; s += kv * kv; }
      s += __shfl_xor(s, 16, 32);
      s += __shfl_xor(s,  8, 32);
      s += __shfl_xor(s,  4, 32);
      s += __shfl_xor(s,  2, 32);
      s += __shfl_xor(s,  1, 32);
      if (tid == 0) kkLDS = s;
    }
    __syncthreads();
    if (tid < HALF) {
      float vps = parts[tid] + parts[tid + 128] + parts[tid + 256] + parts[tid + 384];
      dvs[tid] = kLDS[tid] - vps / (kkLDS + 1e-6f);
    }
    __syncthreads();
    // rank-1 update M += dvs (x) k
    const float dv = dvs[row];
#pragma unroll
    for (int j = 0; j < 32; ++j) Mreg[j] += dv * kLDS[cb + j];
    __syncthreads();
  }

  // read with last-position query
  if (tid < HALF) kLDS[tid] = pb[(long)(LL - 1) * HALF + tid];
  __syncthreads();
  float part = 0.f;
#pragma unroll
  for (int j = 0; j < 32; ++j) part += Mreg[j] * kLDS[cb + j];
  parts[tid] = part;
  __syncthreads();
  if (tid < HALF) {
    float cs = parts[tid] + parts[tid + 128] + parts[tid + 256] + parts[tid + 384];
    cs_h[blockIdx.x * HALF + tid] = (_Float16)cs;
  }
}

// ---------------------------------------------------------------------------
// Kernel 3: out[64 x 32000] = cs[64x128] @ Wo + bo, WMMA tiles.
// 8 waves/block, each wave one 16x16 tile; grid covers 4 x 2000 tiles.
// ---------------------------------------------------------------------------
__global__ __launch_bounds__(256) void out_kernel(
    const _Float16* __restrict__ cs_h, const _Float16* __restrict__ Wot,
    const float* __restrict__ bo, float* __restrict__ out) {
  const int tid  = threadIdx.x;
  const int lane = tid & 31;
  const int w    = tid >> 5;
  const long tile = (long)blockIdx.x * 8 + w;     // 0..7999
  const int nt = (int)(tile >> 2);                // 0..1999
  const int mt = (int)(tile & 3);                 // 0..3

  v8f acc = {};
#pragma unroll
  for (int k0 = 0; k0 < HALF; k0 += 32) {
    v16h a = frag_ld(cs_h + (long)mt * 16 * HALF + k0, HALF, lane);
    v16h b = frag_ld(Wot + (long)nt * 16 * HALF + k0, HALF, lane);
    acc = wmma_f16(a, b, acc);
  }
  const int lo = lane & 15, hi = lane >> 4;
  const int n = nt * 16 + lo;
  const float bias = bo[n];
#pragma unroll
  for (int g = 0; g < 8; ++g) {
    const int r = mt * 16 + g + hi * 8;
    out[(long)r * VV + n] = acc[g] + bias;
  }
}

// ---------------------------------------------------------------------------
extern "C" void kernel_launch(void* const* d_in, const int* in_sizes, int n_in,
                              void* d_out, int out_size, void* d_ws, size_t ws_size,
                              hipStream_t stream) {
  const int*   seq   = (const int*)d_in[0];
  const float* embed = (const float*)d_in[1];
  const float* W1    = (const float*)d_in[2];
  const float* b1    = (const float*)d_in[3];
  const float* W2    = (const float*)d_in[4];
  const float* b2    = (const float*)d_in[5];
  const float* gamma = (const float*)d_in[6];
  const float* beta  = (const float*)d_in[7];
  const float* Wp    = (const float*)d_in[8];
  const float* bp    = (const float*)d_in[9];
  const float* Wo    = (const float*)d_in[10];
  const float* bo    = (const float*)d_in[11];
  float* out = (float*)d_out;

  // workspace carve-up (256B aligned)
  char* ws = (char*)d_ws;
  size_t off = 0;
  auto carve = [&](size_t bytes) -> void* {
    void* p = ws + off;
    off = (off + bytes + 255) & ~(size_t)255;
    return p;
  };
  _Float16* W1t  = (_Float16*)carve((size_t)H  * H2 * 2);     // [512][256]
  _Float16* W2t  = (_Float16*)carve((size_t)H2 * H  * 2);     // [256][512]
  _Float16* Wpt  = (_Float16*)carve((size_t)H  * HALF * 2);   // [128][256]
  _Float16* Wot  = (_Float16*)carve((size_t)HALF * VV * 2);   // [32000][128]
  float*    p_ws = (float*)   carve((size_t)BB * LL * HALF * 4);
  _Float16* cs_h = (_Float16*)carve((size_t)BB * HALF * 2);
  (void)ws_size; (void)in_sizes; (void)n_in; (void)out_size;

  // 0) one-shot weight convert+transpose to f16 N-major
  {
    long n;
    n = (long)H * H2;
    cvt_t_kernel<<<dim3((n + 255) / 256), dim3(256), 0, stream>>>(W1, W1t, H, H2);
    n = (long)H2 * H;
    cvt_t_kernel<<<dim3((n + 255) / 256), dim3(256), 0, stream>>>(W2, W2t, H2, H);
    n = (long)H * HALF;
    cvt_t_kernel<<<dim3((n + 255) / 256), dim3(256), 0, stream>>>(Wp, Wpt, H, HALF);
    n = (long)HALF * VV;
    cvt_t_kernel<<<dim3((n + 255) / 256), dim3(256), 0, stream>>>(Wo, Wot, HALF, VV);
  }

  // 1) encoder + projection: 131072 tokens / 16 per block
  encoder_kernel<<<dim3((BB * LL) / TM), dim3(256), 0, stream>>>(
      seq, embed, W1t, b1, W2t, b2, gamma, beta, Wpt, bp, p_ws);

  // 2) sequential delta-rule recurrence, one block per batch element
  delta_kernel<<<dim3(BB), dim3(512), 0, stream>>>(p_ws, cs_h);

  // 3) vocab projection
  out_kernel<<<dim3((4 * (VV / 16)) / 8), dim3(256), 0, stream>>>(cs_h, Wot, bo, out);
}
